// BiLSTM_CRF_6794638262741
// MI455X (gfx1250) — compile-verified, matrix operand order checked by reading, not run on hardware
//
#include <hip/hip_runtime.h>
#include <hip/hip_bf16.h>
#include <stdint.h>

// ---------------------------------------------------------------------------
// BiLSTM-CRF for MI455X (gfx1250, wave32, WMMA bf16 16x16x32)
//   V=50257, E=128, H=128, T=12, B=64, S=512
// Pipeline:
//   k_cvt    : weights f32 -> bf16 (W_ih/W_hh both dirs, W_out padded N->16)
//   k_gather : embedding gather -> xemb[s][b][e] bf16
//   k_lstm   : 8 persistent WGs (2 dirs x 4 batch-tiles of 16). W_ih+W_hh
//              staged to LDS via TDM tensor_load_to_lds (fallback: copy loop).
//              Strided N-tile mapping keeps i/f/g/o gates for one (row,k) in
//              one lane -> activations fully register-resident, no gate LDS.
//              kc-outer loop nest: one live A fragment -> no scratch spills.
//   k_emis   : emissions = [h_f ; h_b reversed] @ W_out^T + b_out (WMMA)
//   k_crf    : Viterbi forward + backtrace, 1 wave per batch element
// ---------------------------------------------------------------------------

typedef __attribute__((ext_vector_type(16))) __bf16        v16bf;
typedef __attribute__((ext_vector_type(8)))  float         v8f;
typedef __attribute__((ext_vector_type(4)))  unsigned int  v4u;
typedef __attribute__((ext_vector_type(8)))  int           v8i;
typedef __attribute__((ext_vector_type(4)))  int           v4i;

constexpr int E  = 128;
constexpr int Hh = 128;
constexpr int T  = 12;
constexpr int Bb = 64;
constexpr int S  = 512;
constexpr int G4 = 4 * Hh;   // 512 gates

#if defined(__gfx1250__) && __has_builtin(__builtin_amdgcn_tensor_load_to_lds) && \
    __has_builtin(__builtin_amdgcn_s_wait_tensorcnt)
#define USE_TDM 1
#else
#define USE_TDM 0
#endif

__device__ __forceinline__ __bf16 f2bf(float f) {
  unsigned u = __builtin_bit_cast(unsigned, f);
  unsigned r = u + 0x7FFFu + ((u >> 16) & 1u);           // round-to-nearest-even
  unsigned short h = (unsigned short)(r >> 16);
  return __builtin_bit_cast(__bf16, h);
}

// Load a 16x32 bf16 A/B fragment (ISA 7.12.2 16-bit layout).
// A: base points at tile origin A[r][k], ld = row stride (elements).
// B (K x N, stored as W[n][k] row-major): identical mapping with r == n.
__device__ __forceinline__ v16bf frag16(const __bf16* base, int ld) {
  const int lane = threadIdx.x & 31;
  const int r = lane & 15, hv = lane >> 4;
  union { unsigned u[8]; v16bf v; } f;
#pragma unroll
  for (int i = 0; i < 8; ++i) {
    const int kb = ((i >> 2) << 4) + (hv << 3) + ((i & 3) << 1); // even K offset
    f.u[i] = *(const unsigned*)(base + r * ld + kb);             // 2 x bf16
  }
  return f.v;
}

__device__ __forceinline__ v8f wmma_bf16(v16bf a, v16bf b, v8f c) {
  return __builtin_amdgcn_wmma_f32_16x16x32_bf16(false, a, false, b,
                                                 (short)0, c, false, false);
}

__device__ __forceinline__ float sigf(float x) {
  return 1.0f / (1.0f + __expf(-x));
}

#if USE_TDM
// Tensor Data Mover: DMA a 512x128 2-byte-element 2D tile (row-major,
// stride 128) from global memory to LDS. D# per CDNA5 ISA.
__device__ __forceinline__ void tdm_load_2d(unsigned lds_off, const void* gptr) {
  const unsigned long long ga = (unsigned long long)(uintptr_t)gptr;
  v4u g0;
  g0[0] = 1u;                                         // count=1, no gather
  g0[1] = lds_off;                                    // lds_addr (bytes)
  g0[2] = (unsigned)ga;                               // global_addr[31:0]
  g0[3] = (unsigned)((ga >> 32) & 0x01FFFFFFu)        // global_addr[56:32]
        | 0x80000000u;                                // type=2 ("image")
  v8i g1;
  g1[0] = (int)(1u << 16);        // workgroup_mask=0, data_size=1 (2 bytes)
  g1[1] = (int)(128u << 16);      // tensor_dim0[15:0]=128 (bits 63:48)
  g1[2] = (int)(512u << 16);      // tensor_dim1[15:0]=512 (bits 95:80)
  g1[3] = (int)(128u << 16);      // tile_dim0=128 (bits 127:112)
  g1[4] = (int)512;               // tile_dim1=512, tile_dim2=0
  g1[5] = (int)128;               // tensor_dim0_stride=128
  g1[6] = 0;                      // stride hi / dim1_stride lo
  g1[7] = 0;                      // dim1_stride hi
  v4i g2 = {0, 0, 0, 0};
  v4i g3 = {0, 0, 0, 0};
#if __clang_major__ >= 23
  v8i g4 = {0, 0, 0, 0, 0, 0, 0, 0};
  __builtin_amdgcn_tensor_load_to_lds(g0, g1, g2, g3, g4, 0);
#else
  __builtin_amdgcn_tensor_load_to_lds(g0, g1, g2, g3, 0);
#endif
}
#endif

// ---------------------------------------------------------------------------
// Weight conversion: W_ih_f/W_hh_f/W_ih_b/W_hh_b (each 512x128), W_out padded
// to [16][256] bf16 (rows 12..15 zero).
// ---------------------------------------------------------------------------
__global__ void k_cvt(const float* wihf, const float* whhf,
                      const float* wihb, const float* whhb,
                      const float* wout,
                      __bf16* d_wihf, __bf16* d_whhf,
                      __bf16* d_wihb, __bf16* d_whhb, __bf16* d_wop) {
  const int i = blockIdx.x * blockDim.x + threadIdx.x;
  const int W = G4 * E;                       // 65536
  if (i < W)             d_wihf[i]           = f2bf(wihf[i]);
  else if (i < 2 * W)    d_whhf[i - W]       = f2bf(whhf[i - W]);
  else if (i < 3 * W)    d_wihb[i - 2 * W]   = f2bf(wihb[i - 2 * W]);
  else if (i < 4 * W)    d_whhb[i - 3 * W]   = f2bf(whhb[i - 3 * W]);
  else if (i < 4 * W + 16 * 2 * Hh) {
    const int j = i - 4 * W;
    const int n = j >> 8, k = j & 255;
    d_wop[j] = (n < T) ? f2bf(wout[n * 2 * Hh + k]) : f2bf(0.0f);
  }
}

// ---------------------------------------------------------------------------
// Embedding gather: xemb[(s*B+b)*E + e] = bf16(emb[x[b][s]][e])
// ---------------------------------------------------------------------------
__global__ void k_gather(const int* x, const float* emb, __bf16* xemb) {
  const size_t i = (size_t)blockIdx.x * blockDim.x + threadIdx.x;
  if (i >= (size_t)S * Bb * E) return;
  const int e = (int)(i & (E - 1));
  const size_t row = i >> 7;                 // s*B + b
  const int b = (int)(row & (Bb - 1));
  const int s = (int)(row >> 6);
  const int tok = x[b * S + s];
  xemb[i] = f2bf(emb[(size_t)tok * E + e]);
}

// ---------------------------------------------------------------------------
// Persistent LSTM recurrence. grid = 8 blocks (dir*4 + btile), 256 threads.
// Dynamic LDS: W_ih[512][128] bf16 | W_hh[512][128] bf16 | h[16][128] bf16
// (total 266,240 B; CDNA5: 320KB/WGP).
// Wave w handles N-tiles {w, w+8, w+16, w+24}  ->  its four C fragments are
// exactly i/f/g/o for hidden slice k = w*16 + lane%16: activations stay in
// registers (no gate LDS, no cross-wave gather on the serial path).
// ---------------------------------------------------------------------------
__global__ void __launch_bounds__(256)
k_lstm(const __bf16* xemb,
       const __bf16* wihf_bf, const __bf16* whhf_bf,
       const __bf16* wihb_bf, const __bf16* whhb_bf,
       const float* b_f, const float* b_b,
       __bf16* hout) {
  extern __shared__ char smem[];
  __bf16* wih = (__bf16*)smem;                  // 512*128
  __bf16* whh = wih + G4 * Hh;                  // 512*128
  __bf16* hld = whh + G4 * Hh;                  // 16*128

  const int tid  = threadIdx.x;
  const int lane = tid & 31;
  const int wv   = tid >> 5;                    // wave 0..7
  const int dir   = blockIdx.x >> 2;
  const int btile = blockIdx.x & 3;
  const int b0    = btile * 16;

  const __bf16* wih_g = dir ? wihb_bf : wihf_bf;
  const __bf16* whh_g = dir ? whhb_bf : whhf_bf;
  const float*  bias  = dir ? b_b     : b_f;

  // Stage weights into LDS.
#if USE_TDM
  if (wv == 0) {
    tdm_load_2d((unsigned)(uintptr_t)wih, wih_g);
    tdm_load_2d((unsigned)(uintptr_t)whh, whh_g);
    __builtin_amdgcn_s_wait_tensorcnt((short)0);
  }
#else
  {
    const uint32_t* s0 = (const uint32_t*)wih_g;
    const uint32_t* s1 = (const uint32_t*)whh_g;
    uint32_t* d0 = (uint32_t*)wih;
    uint32_t* d1 = (uint32_t*)whh;
    for (int i = tid; i < G4 * Hh / 2; i += 256) { d0[i] = s0[i]; d1[i] = s1[i]; }
  }
#endif
  {
    uint32_t* hz = (uint32_t*)hld;
    for (int i = tid; i < 16 * Hh / 2; i += 256) hz[i] = 0u;
  }
  __syncthreads();

  const int n  = lane & 15;
  const int hv = lane >> 4;
  const int k  = wv * 16 + n;                   // this wave's hidden slice

  // Hoist per-tile biases out of the serial loop (tiles nt = j*8 + wv).
  float bvj[4];
#pragma unroll
  for (int j = 0; j < 4; ++j) bvj[j] = bias[(j * 8 + wv) * 16 + n];

  float creg[8];
#pragma unroll
  for (int i = 0; i < 8; ++i) creg[i] = 0.0f;

  for (int s = 0; s < S; ++s) {
    const int s_eff = dir ? (S - 1 - s) : s;
    const __bf16* abase_x = xemb + ((size_t)s_eff * Bb + b0) * E;

    // Prefetch next step's embedding tile rows (global_prefetch_b8).
    if (s + 1 < S) {
      const int s_nxt = dir ? (S - 2 - s) : (s + 1);
      __builtin_prefetch(xemb + ((size_t)s_nxt * Bb + b0 + (tid & 15)) * E, 0, 1);
    }

    // acc[0]=i, acc[1]=f, acc[2]=g, acc[3]=o  (N-tiles wv, wv+8, wv+16, wv+24)
    v8f acc[4];
#pragma unroll
    for (int j = 0; j < 4; ++j)
#pragma unroll
      for (int v = 0; v < 8; ++v) acc[j][v] = bvj[j];

    // kc-outer: one live A fragment, reused by 4 tiles -> low VGPR pressure.
#pragma unroll
    for (int kc = 0; kc < 4; ++kc) {
      const v16bf a = frag16(abase_x + kc * 32, E);
#pragma unroll
      for (int j = 0; j < 4; ++j) {
        const int nt = j * 8 + wv;
        v16bf bw = frag16(wih + nt * 16 * Hh + kc * 32, Hh);
        acc[j] = wmma_bf16(a, bw, acc[j]);
      }
    }
#pragma unroll
    for (int kc = 0; kc < 4; ++kc) {
      const v16bf a = frag16(hld + kc * 32, Hh);
#pragma unroll
      for (int j = 0; j < 4; ++j) {
        const int nt = j * 8 + wv;
        v16bf bw = frag16(whh + nt * 16 * Hh + kc * 32, Hh);
        acc[j] = wmma_bf16(a, bw, acc[j]);
      }
    }
    __syncthreads();   // all hld reads complete before this step's writes

    // Register-resident activations: C layout VGPR v -> M = v + 8*hv, N = n.
#pragma unroll
    for (int v = 0; v < 8; ++v) {
      const float c = sigf(acc[1][v]) * creg[v] + sigf(acc[0][v]) * tanhf(acc[2][v]);
      creg[v] = c;
      const float h = sigf(acc[3][v]) * tanhf(c);
      const __bf16 hb = f2bf(h);
      const int m = hv * 8 + v;
      hld[m * Hh + k] = hb;
      hout[(((size_t)dir * S + s) * Bb + (b0 + m)) * Hh + k] = hb;
    }
    __syncthreads();   // h writes visible before next step's fragment loads
  }
}

// ---------------------------------------------------------------------------
// Emissions: em[s][b][t] = [h_f(s) ; h_b(S-1-s)] @ W_out^T + b_out
// M=S*B=32768 (2048 tiles of 16), N=16 (12 valid), K=256 (8 chunks).
// grid = 256 blocks x 256 threads (8 waves, 1 M-tile each).
// ---------------------------------------------------------------------------
__global__ void __launch_bounds__(256)
k_emis(const __bf16* hout, const __bf16* wop,
       const float* b_out, float* em) {
  const int tid  = threadIdx.x;
  const int lane = tid & 31;
  const int wv   = tid >> 5;
  const int mtile = blockIdx.x * 8 + wv;       // 0..2047
  const int s   = mtile >> 2;                  // 4 tiles per s
  const int b0m = (mtile & 3) << 4;
  const int n  = lane & 15;
  const int hv = lane >> 4;

  v8f acc;
  const float bv = (n < T) ? b_out[n] : 0.0f;
#pragma unroll
  for (int v = 0; v < 8; ++v) acc[v] = bv;

  const __bf16* hf_base = hout + ((size_t)s * Bb + b0m) * Hh;
  const __bf16* hb_base = hout + (size_t)S * Bb * Hh
                               + ((size_t)(S - 1 - s) * Bb + b0m) * Hh;
#pragma unroll
  for (int kc = 0; kc < 4; ++kc) {
    v16bf a  = frag16(hf_base + kc * 32, Hh);
    v16bf bw = frag16(wop + kc * 32, 2 * Hh);           // W_out_pad[n][k]
    acc = wmma_bf16(a, bw, acc);
  }
#pragma unroll
  for (int kc = 0; kc < 4; ++kc) {
    v16bf a  = frag16(hb_base + kc * 32, Hh);
    v16bf bw = frag16(wop + Hh + kc * 32, 2 * Hh);
    acc = wmma_bf16(a, bw, acc);
  }

  if (n < T) {
#pragma unroll
    for (int v = 0; v < 8; ++v) {
      const int b = b0m + hv * 8 + v;
      em[((size_t)s * Bb + b) * T + n] = acc[v];
    }
  }
}

// ---------------------------------------------------------------------------
// Viterbi + backtrace. 64 blocks (one per batch element), 1 wave each.
// out: path [B][S] as float, then best_score [B].
// ---------------------------------------------------------------------------
__global__ void k_crf(const float* em, const unsigned char* mask,
                      const float* start, const float* end, const float* trans,
                      int* hist, float* out) {
  const int b = blockIdx.x;
  const int t = threadIdx.x;                   // 0..31, lanes < T active
  __shared__ float ssc[16];

  float score = 0.0f;
  if (t < T) score = start[t] + em[(size_t)b * T + t];

  for (int s = 1; s < S; ++s) {
    if (t < T) ssc[t] = score;
    __syncthreads();
    if (t < T) {
      float best = -3.4e38f; int bi = 0;
#pragma unroll
      for (int p = 0; p < T; ++p) {
        const float v = ssc[p] + trans[p * T + t];
        if (v > best) { best = v; bi = p; }
      }
      hist[((size_t)(s - 1) * Bb + b) * T + t] = bi;
      if (mask[b * S + s])
        score = best + em[((size_t)s * Bb + b) * T + t];
    }
    __syncthreads();
  }

  if (t < T) ssc[t] = score + end[t];
  __syncthreads();

  if (t == 0) {
    float best = -3.4e38f; int tag = 0;
#pragma unroll
    for (int p = 0; p < T; ++p)
      if (ssc[p] > best) { best = ssc[p]; tag = p; }
    out[(size_t)Bb * S + b] = best;
    out[(size_t)b * S + (S - 1)] = (float)tag;
    for (int s = S - 2; s >= 0; --s) {
      const int prev = hist[((size_t)s * Bb + b) * T + tag];
      if (mask[b * S + s + 1]) tag = prev;
      out[(size_t)b * S + s] = (float)tag;
    }
  }
}

// ---------------------------------------------------------------------------
extern "C" void kernel_launch(void* const* d_in, const int* in_sizes, int n_in,
                              void* d_out, int out_size, void* d_ws, size_t ws_size,
                              hipStream_t stream) {
  (void)in_sizes; (void)n_in; (void)out_size; (void)ws_size;

  const int*           x     = (const int*)d_in[0];
  const unsigned char* mask  = (const unsigned char*)d_in[1];
  const float*         emb   = (const float*)d_in[2];
  const float*         wihf  = (const float*)d_in[3];
  const float*         whhf  = (const float*)d_in[4];
  const float*         b_f   = (const float*)d_in[5];
  const float*         wihb  = (const float*)d_in[6];
  const float*         whhb  = (const float*)d_in[7];
  const float*         b_b   = (const float*)d_in[8];
  const float*         wout  = (const float*)d_in[9];
  const float*         b_out = (const float*)d_in[10];
  const float*         start = (const float*)d_in[11];
  const float*         endv  = (const float*)d_in[12];
  const float*         trans = (const float*)d_in[13];
  float* out = (float*)d_out;

  // Workspace carve-up (256B aligned pieces).
  char* ws = (char*)d_ws;
  auto carve = [&](size_t bytes) {
    char* p = ws;
    ws += (bytes + 255) & ~(size_t)255;
    return p;
  };
  const size_t W = (size_t)G4 * E;                       // 65536 elems
  __bf16* wihf_bf = (__bf16*)carve(W * 2);
  __bf16* whhf_bf = (__bf16*)carve(W * 2);
  __bf16* wihb_bf = (__bf16*)carve(W * 2);
  __bf16* whhb_bf = (__bf16*)carve(W * 2);
  __bf16* wop_bf  = (__bf16*)carve((size_t)16 * 2 * Hh * 2);
  __bf16* xemb    = (__bf16*)carve((size_t)S * Bb * E * 2);
  __bf16* hout    = (__bf16*)carve((size_t)2 * S * Bb * Hh * 2);
  float*  em      = (float*)carve((size_t)S * Bb * T * 4);
  int*    hist    = (int*)carve((size_t)(S - 1) * Bb * T * 4);

  // 1) Convert weights to bf16 (+ padded W_out).
  {
    const int total = 4 * (int)W + 16 * 2 * Hh;
    k_cvt<<<(total + 255) / 256, 256, 0, stream>>>(
        wihf, whhf, wihb, whhb, wout,
        wihf_bf, whhf_bf, wihb_bf, whhb_bf, wop_bf);
  }
  // 2) Embedding gather.
  {
    const size_t total = (size_t)S * Bb * E;
    k_gather<<<(unsigned)((total + 255) / 256), 256, 0, stream>>>(x, emb, xemb);
  }
  // 3) Persistent bidirectional LSTM (8 WGs, 266,240 B LDS each).
  {
    const size_t smem = (size_t)G4 * Hh * 2 * 2    // W_ih + W_hh bf16
                      + (size_t)16 * Hh * 2;       // h bf16
    k_lstm<<<8, 256, smem, stream>>>(xemb, wihf_bf, whhf_bf, wihb_bf, whhb_bf,
                                     b_f, b_b, hout);
  }
  // 4) Emissions.
  k_emis<<<256, 256, 0, stream>>>(hout, wop_bf, b_out, em);
  // 5) Viterbi decode + backtrace.
  k_crf<<<Bb, 32, 0, stream>>>(em, mask, start, endv, trans, hist, out);
}
